// GATv2Model_26207890440614
// MI455X (gfx1250) — matrix operand order repaired and checked by reading, not compiled, not run on hardware
//
#include <hip/hip_runtime.h>
#include <hip/hip_bf16.h>

// ---------------------------------------------------------------------------
// GATv2 pipeline for MI455X (gfx1250, wave32, WMMA).
// Dense GEMMs: v_wmma_f32_16x16x32_f16, f32 accumulate.
//   - A operands: branch-free float4 (b128) loads, row-clamped tails.
//   - B operands: weights pre-packed once into the per-lane WMMA-B f16
//     layout -> one contiguous 32B load per lane, coalesced wave-wide.
// Attention phase: float4 gathers + wave-coalesced f32 atomics that stay
// resident in the 192MB L2.
// ---------------------------------------------------------------------------

typedef __attribute__((ext_vector_type(16))) _Float16 v16h;
typedef __attribute__((ext_vector_type(8)))  float    v8f;

#define NEG_SLOPE 0.2f
#define BN_EPS    1e-5f

__device__ __forceinline__ v16h cvt_frag(float4 x0, float4 x1, float4 x2, float4 x3) {
  v16h a;
  a[0]  = (_Float16)x0.x; a[1]  = (_Float16)x0.y; a[2]  = (_Float16)x0.z; a[3]  = (_Float16)x0.w;
  a[4]  = (_Float16)x1.x; a[5]  = (_Float16)x1.y; a[6]  = (_Float16)x1.z; a[7]  = (_Float16)x1.w;
  a[8]  = (_Float16)x2.x; a[9]  = (_Float16)x2.y; a[10] = (_Float16)x2.z; a[11] = (_Float16)x2.w;
  a[12] = (_Float16)x3.x; a[13] = (_Float16)x3.y; a[14] = (_Float16)x3.z; a[15] = (_Float16)x3.w;
  return a;
}

// ---- weight packer: W[K,NC] f32 -> WMMA-B f16 per-lane layout --------------
__global__ void k_pack_w(const float* __restrict__ W, _Float16* __restrict__ out,
                         int K, int NC, long total) {
  long idx = (long)blockIdx.x * blockDim.x + threadIdx.x;
  if (idx >= total) return;
  int j    = (int)(idx & 15);
  int lane = (int)((idx >> 4) & 31);
  long rest = idx >> 9;
  int NCT  = NC >> 4;
  int tile = (int)(rest % NCT);
  int ks   = (int)(rest / NCT);
  int c = tile * 16 + (lane & 15);
  int k = ks * 32 + (lane >> 4) * 8 + (j & 7) + (j >> 3) * 16;
  out[idx] = (k < K) ? (_Float16)W[(long)k * NC + c] : (_Float16)0.0f;
}

__global__ void k_zero(float* __restrict__ p, long n) {
  long i = (long)blockIdx.x * blockDim.x + threadIdx.x;
  long stride = (long)gridDim.x * blockDim.x;
  for (; i < n; i += stride) p[i] = 0.0f;
}

// Generic WMMA GEMM: out[M,NC] = A[M,KK] @ W (+bias) (+addend).
// grid = (ceil(M/128), NC/64), block = 256 (8 waves, one 16x64 tile each).
template <int KK, int LDA, int LDO, int NCT, bool BIAS, bool ADD>
__global__ void k_gemm(const float* __restrict__ A,
                       const _Float16* __restrict__ Wp,
                       const float* __restrict__ bias,
                       const float* __restrict__ addend, int add_ld,
                       float* __restrict__ out, long M) {
  const int lane = threadIdx.x & 31;
  const int wave = threadIdx.x >> 5;
  const long row0 = (long)blockIdx.x * 128 + wave * 16;
  if (row0 >= M) return;
  const int tb = blockIdx.y * 4;

  long r = row0 + (lane & 15);
  if (r > M - 1) r = M - 1;                       // clamp: always in bounds
  const float* arow = A + r * (long)LDA + (lane >> 4) * 8;
  const _Float16* wbase = Wp + (long)tb * 512 + (long)lane * 16;

  v8f acc[4] = { {}, {}, {}, {} };
  for (int k0 = 0; k0 < KK; k0 += 32) {
    const float4* ap = (const float4*)(arow + k0);
    v16h a = cvt_frag(ap[0], ap[1], ap[4], ap[5]);
    const _Float16* wk = wbase + (long)(k0 >> 5) * (NCT * 512);
#pragma unroll
    for (int nt = 0; nt < 4; ++nt) {
      v16h b = *(const v16h*)(wk + nt * 512);
      acc[nt] = __builtin_amdgcn_wmma_f32_16x16x32_f16(
          false, a, false, b, (short)0, acc[nt], false, false);
    }
  }

  const int rowadd = (lane >> 4) * 8;
  if (row0 + 16 <= M) {                           // full tile: branch-free
#pragma unroll
    for (int nt = 0; nt < 4; ++nt) {
      int col = (tb + nt) * 16 + (lane & 15);
      float bv = BIAS ? bias[col] : 0.0f;
#pragma unroll
      for (int vr = 0; vr < 8; ++vr) {
        long rr = row0 + vr + rowadd;
        float v = acc[nt][vr] + bv;
        if (ADD) v += addend[rr * (long)add_ld + col];
        out[rr * (long)LDO + col] = v;
      }
    }
  } else {
#pragma unroll
    for (int nt = 0; nt < 4; ++nt) {
      int col = (tb + nt) * 16 + (lane & 15);
      float bv = BIAS ? bias[col] : 0.0f;
#pragma unroll
      for (int vr = 0; vr < 8; ++vr) {
        long rr = row0 + vr + rowadd;
        if (rr < M) {
          float v = acc[nt][vr] + bv;
          if (ADD) v += addend[rr * (long)add_ld + col];
          out[rr * (long)LDO + col] = v;
        }
      }
    }
  }
}

// Edge projection + scatter-sum: s[col[e],:] += edge_attr[e,:]@W_ep + b_ep,
// cnt[col[e]] += 1.  16 edges per wave, K=16 zero-padded to 32 (compile-time).
__global__ void k_edge_proj(const float* __restrict__ ea,
                            const _Float16* __restrict__ Wp,
                            const float* __restrict__ bias,
                            const int* __restrict__ colidx,
                            float* __restrict__ s, float* __restrict__ cnt,
                            long E) {
  const int lane = threadIdx.x & 31;
  const int wave = threadIdx.x >> 5;
  const long e0 = (long)blockIdx.x * 128 + wave * 16;
  if (e0 >= E) return;

  long e = e0 + (lane & 15);
  if (e > E - 1) e = E - 1;
  const float4* ap = (const float4*)(ea + e * 16 + (lane >> 4) * 8);
  float4 z4 = make_float4(0.f, 0.f, 0.f, 0.f);
  v16h a = cvt_frag(ap[0], ap[1], z4, z4);        // K 16..31 are zero pad

  const _Float16* wbase = Wp + (long)lane * 16;
  v8f acc[4] = { {}, {}, {}, {} };
#pragma unroll
  for (int nt = 0; nt < 4; ++nt) {
    v16h b = *(const v16h*)(wbase + nt * 512);
    acc[nt] = __builtin_amdgcn_wmma_f32_16x16x32_f16(
        false, a, false, b, (short)0, acc[nt], false, false);
  }

  const int rowadd = (lane >> 4) * 8;
  const int cl = lane & 15;
  float bv0 = bias[cl], bv1 = bias[16 + cl], bv2 = bias[32 + cl], bv3 = bias[48 + cl];

  if (e0 + 16 <= E) {                             // full tile: one load per row
#pragma unroll
    for (int vr = 0; vr < 8; ++vr) {
      long ee = e0 + vr + rowadd;
      int node = colidx[ee];
      float* ps = s + (long)node * 64 + cl;
      atomicAdd(ps +  0, acc[0][vr] + bv0);
      atomicAdd(ps + 16, acc[1][vr] + bv1);
      atomicAdd(ps + 32, acc[2][vr] + bv2);
      atomicAdd(ps + 48, acc[3][vr] + bv3);
      if (cl == 0) atomicAdd(&cnt[node], 1.0f);
    }
  } else {
#pragma unroll
    for (int vr = 0; vr < 8; ++vr) {
      long ee = e0 + vr + rowadd;
      if (ee < E) {
        int node = colidx[ee];
        float* ps = s + (long)node * 64 + cl;
        atomicAdd(ps +  0, acc[0][vr] + bv0);
        atomicAdd(ps + 16, acc[1][vr] + bv1);
        atomicAdd(ps + 32, acc[2][vr] + bv2);
        atomicAdd(ps + 48, acc[3][vr] + bv3);
        if (cl == 0) atomicAdd(&cnt[node], 1.0f);
      }
    }
  }
}

// agg = s / clip(cnt,1) written to comb[:,64:128] (float4 streaming)
__global__ void k_agg(const float* __restrict__ s, const float* __restrict__ cnt,
                      float* __restrict__ comb, long N) {
  long i = (long)blockIdx.x * blockDim.x + threadIdx.x;
  if (i >= N * 16) return;
  long n = i >> 4; int c4 = (int)(i & 15);
  float ct = cnt[n]; if (ct < 1.0f) ct = 1.0f;
  float inv = 1.0f / ct;
  float4 v = ((const float4*)(s + n * 64))[c4];
  v.x *= inv; v.y *= inv; v.z *= inv; v.w *= inv;
  ((float4*)(comb + n * 128 + 64))[c4] = v;
}

// per-column sum / sumsq; grid = (C/64, RB), block = 256 (4 row-lanes x 64 cols)
__global__ void k_colstats(const float* __restrict__ in, int ld, long M,
                           float* __restrict__ stats) {
  __shared__ float s1[256], s2[256];
  const int c = blockIdx.x * 64 + (threadIdx.x & 63);
  const int rl = threadIdx.x >> 6;
  float sum = 0.0f, sq = 0.0f;
  for (long r = (long)blockIdx.y * 4 + rl; r < M; r += 4L * gridDim.y) {
    float v = in[r * (long)ld + c];
    sum += v; sq += v * v;
  }
  s1[threadIdx.x] = sum; s2[threadIdx.x] = sq;
  __syncthreads();
  if (rl == 0) {
    int t = threadIdx.x;
    sum = s1[t] + s1[t + 64] + s1[t + 128] + s1[t + 192];
    sq  = s2[t] + s2[t + 64] + s2[t + 128] + s2[t + 192];
    atomicAdd(&stats[c], sum);
    atomicAdd(&stats[512 + c], sq);
  }
}

__global__ void k_finalize_stats(float* __restrict__ stats,
                                 const float* __restrict__ g,
                                 const float* __restrict__ be,
                                 int C, float invM) {
  int c = blockIdx.x * blockDim.x + threadIdx.x;
  if (c >= C) return;
  float mean = stats[c] * invM;
  float var  = stats[512 + c] * invM - mean * mean;
  float sc = g[c] * rsqrtf(var + BN_EPS);
  stats[1024 + c] = sc;
  stats[1536 + c] = be[c] - mean * sc;
}

// float4 streaming BN+ReLU; C and leading dims are multiples of 4
__global__ void k_bn_relu(const float* __restrict__ in, int ld_in,
                          float* __restrict__ out, int ld_out,
                          const float* __restrict__ stats, int C, long M) {
  long i = (long)blockIdx.x * blockDim.x + threadIdx.x;
  int C4 = C >> 2;
  if (i >= M * (long)C4) return;
  long r = i / C4; int c4 = (int)(i % C4);
  float4 v  = ((const float4*)(in + r * (long)ld_in))[c4];
  float4 sc = ((const float4*)(stats + 1024))[c4];
  float4 sh = ((const float4*)(stats + 1536))[c4];
  float4 o;
  o.x = fmaxf(v.x * sc.x + sh.x, 0.0f);
  o.y = fmaxf(v.y * sc.y + sh.y, 0.0f);
  o.z = fmaxf(v.z * sc.z + sh.z, 0.0f);
  o.w = fmaxf(v.w * sc.w + sh.w, 0.0f);
  ((float4*)(out + r * (long)ld_out))[c4] = o;
}

// Monotonic uint encoding so atomicMax(uint) == float max.
__device__ __forceinline__ unsigned enc_f(float x) {
  unsigned u = __float_as_uint(x);
  return (u & 0x80000000u) ? ~u : (u | 0x80000000u);
}
__device__ __forceinline__ float dec_f(unsigned u) {
  u = (u & 0x80000000u) ? (u & 0x7fffffffu) : ~u;
  return __uint_as_float(u);
}

// scores for all E+N entries (self-loops appended), per head; float4 gathers
__global__ void k_score(const float* __restrict__ xl, const float* __restrict__ xr,
                        const float* __restrict__ att,
                        const int* __restrict__ rowi, const int* __restrict__ coli,
                        float* __restrict__ score, unsigned* __restrict__ smax,
                        long E, long N) {
  __shared__ __align__(16) float att_s[512];
  for (int i = threadIdx.x; i < 512; i += blockDim.x) att_s[i] = att[i];
  __syncthreads();
  long idx = (long)blockIdx.x * blockDim.x + threadIdx.x;
  long tot = (E + N) * 8;
  if (idx >= tot) return;
  long e = idx >> 3; int h = (int)(idx & 7);
  int src = (e < E) ? rowi[e] : (int)(e - E);
  int dst = (e < E) ? coli[e] : (int)(e - E);
  const float4* pl = (const float4*)(xl + (long)src * 512 + h * 64);
  const float4* pr = (const float4*)(xr + (long)dst * 512 + h * 64);
  const float4* pa = (const float4*)(att_s + h * 64);
  float sc = 0.0f;
#pragma unroll
  for (int i = 0; i < 16; ++i) {
    float4 l = pl[i], r = pr[i], a = pa[i];
    float v;
    v = l.x + r.x; v = v > 0.0f ? v : NEG_SLOPE * v; sc += v * a.x;
    v = l.y + r.y; v = v > 0.0f ? v : NEG_SLOPE * v; sc += v * a.y;
    v = l.z + r.z; v = v > 0.0f ? v : NEG_SLOPE * v; sc += v * a.z;
    v = l.w + r.w; v = v > 0.0f ? v : NEG_SLOPE * v; sc += v * a.w;
  }
  score[idx] = sc;
  atomicMax(&smax[(long)dst * 8 + h], enc_f(sc));
}

__global__ void k_softmax_num(float* __restrict__ score,
                              const unsigned* __restrict__ smax,
                              float* __restrict__ denom,
                              const int* __restrict__ coli, long E, long N) {
  long idx = (long)blockIdx.x * blockDim.x + threadIdx.x;
  long tot = (E + N) * 8;
  if (idx >= tot) return;
  long e = idx >> 3; int h = (int)(idx & 7);
  int dst = (e < E) ? coli[e] : (int)(e - E);
  float m = dec_f(smax[(long)dst * 8 + h]);
  float ex = __expf(score[idx] - m);
  score[idx] = ex;
  atomicAdd(&denom[(long)dst * 8 + h], ex);
}

// wave-per-entry: 32 lanes hit contiguous dst addresses -> coalesced L2 atomics
__global__ void k_scatter_out(const float* __restrict__ score,
                              const float* __restrict__ denom,
                              const float* __restrict__ xl,
                              const int* __restrict__ rowi,
                              const int* __restrict__ coli,
                              float* __restrict__ gatout, long E, long N) {
  const int lane = threadIdx.x & 31;
  const int wave = threadIdx.x >> 5;
  long e = (long)blockIdx.x * 8 + wave;
  long tot = E + N;
  if (e >= tot) return;
  int src = (e < E) ? rowi[e] : (int)(e - E);
  int dst = (e < E) ? coli[e] : (int)(e - E);
  const float* sc = score + e * 8;
  const float* dn = denom + (long)dst * 8;
  const float4* xv = (const float4*)(xl + (long)src * 512);
  float* po = gatout + (long)dst * 512;
#pragma unroll
  for (int i = 0; i < 4; ++i) {
    int c0 = lane * 4 + 128 * i;
    int h = c0 >> 6;
    float alpha = sc[h] / dn[h];
    float4 v = xv[c0 >> 2];
    atomicAdd(&po[c0 + 0], alpha * v.x);
    atomicAdd(&po[c0 + 1], alpha * v.y);
    atomicAdd(&po[c0 + 2], alpha * v.z);
    atomicAdd(&po[c0 + 3], alpha * v.w);
  }
}

// fused BN(p) + ReLU + final matvec with W_p2 [64,1] (float4 loads)
__global__ void k_final(const float* __restrict__ h1,
                        const float* __restrict__ stats,
                        const float* __restrict__ W_p2,
                        const float* __restrict__ b_p2,
                        float* __restrict__ out, long N) {
  long n = (long)blockIdx.x * blockDim.x + threadIdx.x;
  if (n >= N) return;
  const float4* hv = (const float4*)(h1 + n * 64);
  const float4* scv = (const float4*)(stats + 1024);
  const float4* shv = (const float4*)(stats + 1536);
  const float4* wv  = (const float4*)W_p2;
  float acc = 0.0f;
#pragma unroll
  for (int i = 0; i < 16; ++i) {
    float4 v = hv[i], sc = scv[i], sh = shv[i], w = wv[i];
    acc += fmaxf(v.x * sc.x + sh.x, 0.0f) * w.x;
    acc += fmaxf(v.y * sc.y + sh.y, 0.0f) * w.y;
    acc += fmaxf(v.z * sc.z + sh.z, 0.0f) * w.z;
    acc += fmaxf(v.w * sc.w + sh.w, 0.0f) * w.w;
  }
  out[n] = acc + b_p2[0];
}

// ---------------------------------------------------------------------------
extern "C" void kernel_launch(void* const* d_in, const int* in_sizes, int n_in,
                              void* d_out, int out_size, void* d_ws, size_t ws_size,
                              hipStream_t stream) {
  const float* x       = (const float*)d_in[0];
  const int*   ei      = (const int*)d_in[1];
  const float* ea      = (const float*)d_in[2];
  const float* W_ap    = (const float*)d_in[3];
  const float* b_ap    = (const float*)d_in[4];
  const float* W_ep    = (const float*)d_in[5];
  const float* b_ep    = (const float*)d_in[6];
  const float* W_msg   = (const float*)d_in[7];
  // b_msg (d_in[8]) annihilated by BN mean subtraction
  const float* g_msg   = (const float*)d_in[9];
  const float* be_msg  = (const float*)d_in[10];
  const float* W_l     = (const float*)d_in[11];
  const float* W_r     = (const float*)d_in[12];
  const float* att     = (const float*)d_in[13];
  // b_gat (d_in[14]) annihilated by BN mean subtraction
  const float* g_bn    = (const float*)d_in[15];
  const float* be_bn   = (const float*)d_in[16];
  const float* W_p1    = (const float*)d_in[17];
  const float* b_p1    = (const float*)d_in[18];
  const float* g_p     = (const float*)d_in[19];
  const float* be_p    = (const float*)d_in[20];
  const float* W_p2    = (const float*)d_in[21];
  const float* b_p2    = (const float*)d_in[22];
  float* out = (float*)d_out;

  const long N = in_sizes[0] / 64;
  const long E = in_sizes[1] / 2;
  const int* rowi = ei;
  const int* coli = ei + E;

  // workspace layout (floats); every region is a multiple of 8 floats (32B)
  float* wsf = (float*)d_ws;
  float*    s_     = wsf;                         // N*64 scatter sums
  float*    cnt_   = s_ + N * 64;                 // N
  float*    t_     = cnt_ + N;                    // N*64  atom+agg
  float*    y_     = t_ + N * 64;                 // N*64  pre-BN msg
  float*    comb_  = y_ + N * 64;                 // N*128 [msg | agg]
  float*    xl_    = comb_ + N * 128;             // N*512 (later o2)
  float*    xr_    = xl_ + N * 512;               // N*512 (later gatout)
  float*    score_ = xr_ + N * 512;               // (E+N)*8
  unsigned* smax_  = (unsigned*)(score_ + (E + N) * 8); // N*8
  float*    denom_ = (float*)(smax_ + N * 8);     // N*8
  float*    h1_    = denom_ + N * 8;              // N*64
  float*    stats_ = h1_ + N * 64;                // 2048 (sum|sq|scale|shift)
  // packed f16 weights (sizes in halfs, all 32B aligned)
  _Float16* pW_ep  = (_Float16*)(stats_ + 2048);  //  1*4*512 = 2048
  _Float16* pW_ap  = pW_ep + 2048;                //  2*4*512 = 4096
  _Float16* pW_msg = pW_ap + 4096;                //  4096
  _Float16* pW_l   = pW_msg + 4096;               //  4*32*512 = 65536
  _Float16* pW_r   = pW_l + 65536;                //  65536
  _Float16* pW_p1  = pW_r + 65536;                // 16*4*512 = 32768
  float*    gatout_ = xr_;                        // alias (xr dead after k_score)
  float*    o2_     = xl_;                        // alias (xl dead after scatter)

  const int B = 256;
  auto nb = [](long n, int b) { return (unsigned)((n + b - 1) / b); };

  // ---- stage 0: init + weight packing
  k_zero<<<2048, B, 0, stream>>>(s_, N * 64 + N);
  k_zero<<<256, B, 0, stream>>>((float*)smax_, N * 8);
  k_zero<<<256, B, 0, stream>>>(denom_, N * 8);
  k_pack_w<<<nb(2048, B),  B, 0, stream>>>(W_ep,  pW_ep,  16,  64, 2048);
  k_pack_w<<<nb(4096, B),  B, 0, stream>>>(W_ap,  pW_ap,  64,  64, 4096);
  k_pack_w<<<nb(4096, B),  B, 0, stream>>>(W_msg, pW_msg, 64,  64, 4096);
  k_pack_w<<<nb(65536, B), B, 0, stream>>>(W_l,   pW_l,   128, 512, 65536);
  k_pack_w<<<nb(65536, B), B, 0, stream>>>(W_r,   pW_r,   128, 512, 65536);
  k_pack_w<<<nb(32768, B), B, 0, stream>>>(W_p1,  pW_p1,  512, 64, 32768);

  // ---- stage 1: edge projection + scatter-mean numerators
  k_edge_proj<<<nb(E, 128), B, 0, stream>>>(ea, pW_ep, b_ep, coli, s_, cnt_, E);
  k_agg<<<nb(N * 16, B), B, 0, stream>>>(s_, cnt_, comb_, N);

  // ---- stage 2: t = x@W_ap + b_ap + agg ; y = t@W_msg (b_msg dropped)
  dim3 g64(nb(N, 128), 1);
  k_gemm<64, 64, 64, 4, true, true><<<g64, B, 0, stream>>>(
      x, pW_ap, b_ap, comb_ + 64, 128, t_, N);
  k_gemm<64, 64, 64, 4, false, false><<<g64, B, 0, stream>>>(
      t_, pW_msg, nullptr, nullptr, 0, y_, N);

  // ---- stage 3: BN(msg) + ReLU -> comb[:,0:64]
  k_zero<<<8, B, 0, stream>>>(stats_, 1024);
  k_colstats<<<dim3(1, 64), B, 0, stream>>>(y_, 64, N, stats_);
  k_finalize_stats<<<1, 64, 0, stream>>>(stats_, g_msg, be_msg, 64, 1.0f / (float)N);
  k_bn_relu<<<nb(N * 16, B), B, 0, stream>>>(y_, 64, comb_, 128, stats_, 64, N);

  // ---- stage 4: xl/xr = comb @ W_l / W_r   (dominant GEMMs)
  dim3 g512(nb(N, 128), 8);
  k_gemm<128, 128, 512, 32, false, false><<<g512, B, 0, stream>>>(
      comb_, pW_l, nullptr, nullptr, 0, xl_, N);
  k_gemm<128, 128, 512, 32, false, false><<<g512, B, 0, stream>>>(
      comb_, pW_r, nullptr, nullptr, 0, xr_, N);

  // ---- stage 5: attention (per-dst softmax over E+N entries)
  long tot = (E + N) * 8;
  k_score<<<nb(tot, B), B, 0, stream>>>(xl_, xr_, att, rowi, coli, score_, smax_, E, N);
  k_zero<<<4096, B, 0, stream>>>(gatout_, N * 512);   // xr dead; reuse as accumulator
  k_softmax_num<<<nb(tot, B), B, 0, stream>>>(score_, smax_, denom_, coli, E, N);
  k_scatter_out<<<nb(E + N, 8), B, 0, stream>>>(score_, denom_, xl_, rowi, coli, gatout_, E, N);

  // ---- stage 6: BN(gat) + ReLU -> o2 (b_gat dropped: pre-BN constant)
  k_zero<<<8, B, 0, stream>>>(stats_, 1024);
  k_colstats<<<dim3(8, 64), B, 0, stream>>>(gatout_, 512, N, stats_);
  k_finalize_stats<<<1, 512, 0, stream>>>(stats_, g_bn, be_bn, 512, 1.0f / (float)N);
  k_bn_relu<<<nb(N * 128, B), B, 0, stream>>>(gatout_, 512, o2_, 512, stats_, 512, N);

  // ---- stage 7: h1 = o2 @ W_p1 + b_p1 ; BN(p) ; final matvec
  k_gemm<512, 512, 64, 4, true, false><<<g64, B, 0, stream>>>(
      o2_, pW_p1, b_p1, nullptr, 0, h1_, N);
  k_zero<<<8, B, 0, stream>>>(stats_, 1024);
  k_colstats<<<dim3(1, 64), B, 0, stream>>>(h1_, 64, N, stats_);
  k_finalize_stats<<<1, 64, 0, stream>>>(stats_, g_p, be_p, 64, 1.0f / (float)N);
  k_final<<<nb(N, B), B, 0, stream>>>(h1_, stats_, W_p2, b_p2, out, N);
}
